// Decoder_1357209666110
// MI455X (gfx1250) — matrix operand order repaired and checked by reading, not compiled, not run on hardware
//
#include <hip/hip_runtime.h>
#include <hip/hip_bf16.h>

// ---------------- types for WMMA ----------------
typedef __attribute__((ext_vector_type(16))) __bf16 v16bf;
typedef __attribute__((ext_vector_type(8)))  float  v8f;

union BfTile { uint4 q[2]; v16bf v; };

static constexpr int V  = 45000;
static constexpr int E  = 300;
static constexpr int H  = 512;
static constexpr int B  = 32;
static constexpr int T  = 32;
static constexpr int S  = 400;
static constexpr int NOOV = 30;
static constexpr int VE = V + NOOV;        // 45030
static constexpr int KP = 320;             // padded E for WMMA K
static constexpr int NTB = 4;              // N-tiles per wave (register blocking)
static constexpr int NP_OUT = 45056;       // W_out rows padded to multiple of 16*NTB
static constexpr int OUT_GROUPS = NP_OUT / (16 * NTB);   // 704
static constexpr float NEG = -1e12f;

__device__ __forceinline__ __bf16 f2bf(float f) {
    union { float f; unsigned u; } in; in.f = f;
    unsigned r = in.u + 0x7FFFu + ((in.u >> 16) & 1u);
    union { unsigned short u; __bf16 b; } out; out.u = (unsigned short)(r >> 16);
    return out.b;
}

__device__ __forceinline__ float sigmoidf_(float x) { return 1.0f / (1.0f + expf(-x)); }

// A fragment: row-major A[16 x K], lda in elements. lane<16: K=k0..k0+7,k0+16..23 ; lane>=16: +8
__device__ __forceinline__ v16bf load_a(const __bf16* A, int lda, int k0, int lane) {
    int m = lane & 15, half = lane >> 4;
    const __bf16* p = A + (size_t)m * lda + k0 + half * 8;
    BfTile t;
    t.q[0] = *(const uint4*)(p);
    t.q[1] = *(const uint4*)(p + 16);
    return t.v;
}
// B fragment from TRANSPOSED storage Bt[N x K] (row n == column n of B). lane<16: K=k0..k0+7,k0+16..23 ; lane>=16: +8
__device__ __forceinline__ v16bf load_b(const __bf16* Bt, int ldb, int k0, int lane) {
    int n = lane & 15, half = lane >> 4;
    const __bf16* p = Bt + (size_t)n * ldb + k0 + half * 8;
    BfTile t;
    t.q[0] = *(const uint4*)(p);
    t.q[1] = *(const uint4*)(p + 16);
    return t.v;
}
__device__ __forceinline__ v8f bf16_wmma(v16bf a, v16bf b, v8f c) {
    return __builtin_amdgcn_wmma_f32_16x16x32_bf16(false, a, false, b, (short)0, c, false, false);
}

// ---------------- one-time conversion kernels ----------------
__global__ void k_cvt(const float* __restrict__ src, __bf16* __restrict__ dst, int n) {
    int i = blockIdx.x * 256 + threadIdx.x;
    if (i < n) dst[i] = f2bf(src[i]);
}

// src f32 [K x N] row-major  ->  dst bf16 [Npad x Kpad] row-major (transposed, zero padded)
__global__ void k_cvt_trans(const float* __restrict__ src, __bf16* __restrict__ dst,
                            int K, int N, int Kpad, int Npad) {
    __shared__ float tile[32][33];
    int n0 = blockIdx.x * 32, k0 = blockIdx.y * 32;
    for (int r = threadIdx.y; r < 32; r += 8) {
        int k = k0 + r, n = n0 + threadIdx.x;
        tile[r][threadIdx.x] = (k < K && n < N) ? src[(size_t)k * N + n] : 0.0f;
    }
    __syncthreads();
    for (int r = threadIdx.y; r < 32; r += 8) {
        int n = n0 + r, k = k0 + threadIdx.x;
        if (n < Npad && k < Kpad) dst[(size_t)n * Kpad + k] = f2bf(tile[threadIdx.x][r]);
    }
}

__global__ void k_init(const float* __restrict__ h0, const float* __restrict__ c0,
                       float* __restrict__ h, float* __restrict__ c, float* __restrict__ ctx) {
    int i = blockIdx.x * 256 + threadIdx.x;
    if (i < B * H) { h[i] = h0[i]; c[i] = c0[i]; ctx[i] = 0.0f; }
}

// software-pipelined 4-way N-blocked WMMA tile loop (double-buffered fragments)
__device__ __forceinline__ void wmma_pipeline(const __bf16* __restrict__ A,
                                              const __bf16* __restrict__ Bt,
                                              int K, v8f acc[NTB], int lane) {
    v16bf a0 = load_a(A, K, 0, lane);
    v16bf b0[NTB];
#pragma unroll
    for (int j = 0; j < NTB; ++j) b0[j] = load_b(Bt + (size_t)j * 16 * K, K, 0, lane);
#pragma unroll
    for (int k0 = 0; k0 < H; k0 += 32) {          // K == H for all pipelined GEMMs
        v16bf a1 = {};
        v16bf b1[NTB] = {};
        if (k0 + 32 < H) {
            a1 = load_a(A, K, k0 + 32, lane);
#pragma unroll
            for (int j = 0; j < NTB; ++j) b1[j] = load_b(Bt + (size_t)j * 16 * K, K, k0 + 32, lane);
        }
#pragma unroll
        for (int j = 0; j < NTB; ++j) acc[j] = bf16_wmma(a0, b0[j], acc[j]);
        a0 = a1;
#pragma unroll
        for (int j = 0; j < NTB; ++j) b0[j] = b1[j];
    }
}

// ---------------- memories = enc_out @ W_enc + b_enc  (WMMA bf16, pipelined, 4-way N blocking) ---------------
__global__ void k_memories(const __bf16* __restrict__ encbf, const __bf16* __restrict__ WencT,
                           const float* __restrict__ b_enc, float* __restrict__ mem) {
    int wglob = blockIdx.x * 8 + (threadIdx.x >> 5);
    int lane = threadIdx.x & 31;
    int mt = wglob >> 3;          // 800 M tiles (B*S=12800)
    int ng = wglob & 7;           // 8 N groups of 4 tiles (H=512)
    if (mt >= 800) return;
    v8f acc[NTB] = {};
    const __bf16* A  = encbf + (size_t)mt * 16 * H;
    const __bf16* Bt = WencT + (size_t)ng * (16 * NTB) * H;
    wmma_pipeline(A, Bt, H, acc, lane);
    int half = lane >> 4, nn = lane & 15;
#pragma unroll
    for (int j = 0; j < NTB; ++j) {
        int n = ng * (16 * NTB) + j * 16 + nn;
        float bias = b_enc[n];
#pragma unroll
        for (int r = 0; r < 8; ++r) {
            int m = mt * 16 + r + 8 * half;
            mem[(size_t)m * H + n] = acc[j][r] + bias;
        }
    }
}

// ---------------- per-step kernels ----------------
// x = [emb(y) | ctx] @ W_red + b_red  -> bf16 padded [B x 320]
__global__ void k_reduce(const int* __restrict__ trg, const float* __restrict__ emb,
                         const float* __restrict__ ctx, const float* __restrict__ W_red,
                         const float* __restrict__ b_red, __bf16* __restrict__ xbf, int t) {
    int idx = blockIdx.x * 256 + threadIdx.x;
    if (idx >= B * KP) return;
    int b = idx / KP, n = idx % KP;
    if (n >= E) { xbf[idx] = f2bf(0.0f); return; }
    int y = trg[b * T + t];
    const float* e  = emb + (size_t)y * E;
    const float* cx = ctx + (size_t)b * H;
    float acc = b_red[n];
    for (int k = 0; k < E; ++k) acc += e[k] * W_red[(size_t)k * E + n];
    for (int k = 0; k < H; ++k) acc += cx[k] * W_red[(size_t)(E + k) * E + n];
    xbf[idx] = f2bf(acc);
}

__global__ void k_h2bf(const float* __restrict__ h, __bf16* __restrict__ hbf) {
    int i = blockIdx.x * 256 + threadIdx.x;
    if (i < B * H) hbf[i] = f2bf(h[i]);
}

// gates = x@W_ih + h@W_hh + b ; fused LSTM elementwise. 32 blocks, one per 16-column slice of H.
__global__ void k_gates(const __bf16* __restrict__ xbf, const __bf16* __restrict__ hbf,
                        const __bf16* __restrict__ WihT, const __bf16* __restrict__ WhhT,
                        const float* __restrict__ b_ih, const float* __restrict__ b_hh,
                        float* __restrict__ c_state, float* __restrict__ h_state) {
    __shared__ float sg[4][B][16];
    int j0 = blockIdx.x * 16;
    int w = threadIdx.x >> 5, lane = threadIdx.x & 31;
    int gate = w >> 1, mt = w & 1;
    int ncol = gate * H + j0;                // row index into transposed weight storage
    v8f acc = {};
    const __bf16* Ax = xbf + (size_t)mt * 16 * KP;
    const __bf16* Bx = WihT + (size_t)ncol * KP;
    for (int k0 = 0; k0 < KP; k0 += 32) {
        v16bf a = load_a(Ax, KP, k0, lane);
        v16bf b = load_b(Bx, KP, k0, lane);
        acc = bf16_wmma(a, b, acc);
    }
    const __bf16* Ah = hbf + (size_t)mt * 16 * H;
    const __bf16* Bh = WhhT + (size_t)ncol * H;
    for (int k0 = 0; k0 < H; k0 += 32) {
        v16bf a = load_a(Ah, H, k0, lane);
        v16bf b = load_b(Bh, H, k0, lane);
        acc = bf16_wmma(a, b, acc);
    }
    int half = lane >> 4, nn = lane & 15;
    int j = j0 + nn;
    float bias = b_ih[gate * H + j] + b_hh[gate * H + j];
#pragma unroll
    for (int r = 0; r < 8; ++r) {
        int m = mt * 16 + r + 8 * half;
        sg[gate][m][nn] = acc[r] + bias;
    }
    __syncthreads();
    for (int p = threadIdx.x; p < B * 16; p += 256) {
        int b = p >> 4, jj = p & 15;
        int jc = j0 + jj;
        float iv = sigmoidf_(sg[0][b][jj]);
        float fv = sigmoidf_(sg[1][b][jj]);
        float gv = tanhf(sg[2][b][jj]);
        float ov = sigmoidf_(sg[3][b][jj]);
        float cn = fv * c_state[b * H + jc] + iv * gv;
        c_state[b * H + jc] = cn;
        h_state[b * H + jc] = ov * tanhf(cn);
    }
}

// attention: energy, mask, softmax, ctx. One block per batch row.
__global__ void k_attn(const float* __restrict__ h, const float* __restrict__ mem,
                       const unsigned char* __restrict__ mask,
                       float* __restrict__ energy_ws, float* __restrict__ ctx) {
    int b = blockIdx.x;
    __shared__ float sh[H];
    __shared__ float se[S];
    __shared__ float sred[256];
    int tid = threadIdx.x;
    for (int k = tid; k < H; k += 256) sh[k] = h[(size_t)b * H + k];
    __syncthreads();
    int wid = tid >> 5, lane = tid & 31;
    for (int s = wid; s < S; s += 8) {
        const float* row = mem + ((size_t)b * S + s) * H;
        float acc = 0.0f;
        int k0 = lane * 16;
        for (int k = k0; k < k0 + 16; k += 4) {
            float4 m4 = *(const float4*)(row + k);
            acc += m4.x * sh[k] + m4.y * sh[k + 1] + m4.z * sh[k + 2] + m4.w * sh[k + 3];
        }
        for (int off = 16; off; off >>= 1) acc += __shfl_xor(acc, off, 32);
        if (lane == 0) {
            float e = mask[b * S + s] ? NEG : acc;
            se[s] = e;
            energy_ws[b * S + s] = e;
        }
    }
    __syncthreads();
    float m = -3.4e38f;
    for (int s = tid; s < S; s += 256) m = fmaxf(m, se[s]);
    sred[tid] = m; __syncthreads();
    for (int st = 128; st; st >>= 1) { if (tid < st) sred[tid] = fmaxf(sred[tid], sred[tid + st]); __syncthreads(); }
    float mx = sred[0]; __syncthreads();
    float sum = 0.0f;
    for (int s = tid; s < S; s += 256) { float ev = expf(se[s] - mx); se[s] = ev; sum += ev; }
    sred[tid] = sum; __syncthreads();
    for (int st = 128; st; st >>= 1) { if (tid < st) sred[tid] += sred[tid + st]; __syncthreads(); }
    float inv = 1.0f / sred[0]; __syncthreads();
    for (int s = tid; s < S; s += 256) se[s] *= inv;
    __syncthreads();
    for (int k = tid; k < H; k += 256) {
        float acc = 0.0f;
        const float* base = mem + (size_t)b * S * H + k;
        for (int s = 0; s < S; ++s) acc += se[s] * base[(size_t)s * H];
        ctx[(size_t)b * H + k] = acc;
    }
}

// li = tanh([h|ctx] @ W_cat + b_cat) -> bf16
__global__ void k_cat(const float* __restrict__ h, const float* __restrict__ ctx,
                      const float* __restrict__ W_cat, const float* __restrict__ b_cat,
                      __bf16* __restrict__ libf) {
    int idx = blockIdx.x * 256 + threadIdx.x;
    if (idx >= B * H) return;
    int b = idx >> 9, n = idx & (H - 1);
    const float* hr = h + (size_t)b * H;
    const float* cr = ctx + (size_t)b * H;
    float acc = b_cat[n];
    for (int k = 0; k < H; ++k) acc += hr[k] * W_cat[(size_t)k * H + n];
    for (int k = 0; k < H; ++k) acc += cr[k] * W_cat[(size_t)(H + k) * H + n];
    libf[idx] = f2bf(tanhf(acc));
}

// logits = li @ W_out + b_out  (WMMA bf16, pipelined, 4-way N register blocking)
__global__ void k_out(const __bf16* __restrict__ libf, const __bf16* __restrict__ WoutT,
                      const float* __restrict__ b_out, float* __restrict__ out, int t) {
    int wglob = blockIdx.x * 8 + (threadIdx.x >> 5);
    int lane = threadIdx.x & 31;
    int group = wglob >> 1;       // N-tile group of 4
    int mt = wglob & 1;
    if (group >= OUT_GROUPS) return;
    v8f acc[NTB] = {};
    const __bf16* A  = libf + (size_t)mt * 16 * H;
    const __bf16* Bt = WoutT + (size_t)group * (16 * NTB) * H;
    wmma_pipeline(A, Bt, H, acc, lane);
    int half = lane >> 4, nn = lane & 15;
#pragma unroll
    for (int j = 0; j < NTB; ++j) {
        int n = group * (16 * NTB) + j * 16 + nn;
        if (n < V) {
            float bias = b_out[n];
#pragma unroll
            for (int r = 0; r < 8; ++r) {
                int bb = mt * 16 + r + 8 * half;
                out[(size_t)(bb * T + t) * VE + n] = acc[j][r] + bias;
            }
        }
    }
}

__global__ void k_zoov(float* __restrict__ out, int t) {
    int idx = blockIdx.x * 256 + threadIdx.x;
    if (idx >= B * NOOV) return;
    int b = idx / NOOV, j = idx % NOOV;
    out[(size_t)(b * T + t) * VE + V + j] = 0.0f;
}

__global__ void k_scatter(const int* __restrict__ ext_src, const float* __restrict__ energy,
                          float* __restrict__ out, int t) {
    int idx = blockIdx.x * 256 + threadIdx.x;
    if (idx >= B * S) return;
    int b = idx / S;
    int col = ext_src[idx];
    atomicAdd(out + (size_t)(b * T + t) * VE + col, energy[idx]);
}

__global__ void k_final(float* __restrict__ out, int t) {
    int idx = blockIdx.x * 256 + threadIdx.x;
    if (idx >= B * VE) return;
    int b = idx / VE, n = idx % VE;
    size_t o = (size_t)(b * T + t) * VE + n;
    float v = out[o];
    if (v == 0.0f) out[o] = NEG;
}

// ---------------- host launcher ----------------
extern "C" void kernel_launch(void* const* d_in, const int* in_sizes, int n_in,
                              void* d_out, int out_size, void* d_ws, size_t ws_size,
                              hipStream_t stream) {
    const int*   trg      = (const int*)d_in[0];
    const int*   ext_src  = (const int*)d_in[1];
    const float* h0       = (const float*)d_in[2];
    const float* c0       = (const float*)d_in[3];
    const float* enc_out  = (const float*)d_in[4];
    const unsigned char* enc_mask = (const unsigned char*)d_in[5];
    const float* embedding = (const float*)d_in[6];
    const float* W_enc = (const float*)d_in[7];
    const float* b_enc = (const float*)d_in[8];
    const float* W_red = (const float*)d_in[9];
    const float* b_red = (const float*)d_in[10];
    const float* W_ih  = (const float*)d_in[11];
    const float* W_hh  = (const float*)d_in[12];
    const float* b_ih  = (const float*)d_in[13];
    const float* b_hh  = (const float*)d_in[14];
    const float* W_cat = (const float*)d_in[15];
    const float* b_cat = (const float*)d_in[16];
    const float* W_out = (const float*)d_in[17];
    const float* b_out = (const float*)d_in[18];
    float* out = (float*)d_out;

    // workspace carve-out (256B aligned regions)
    char* ws = (char*)d_ws;
    size_t off = 0;
    auto carve = [&](size_t bytes) { char* p = ws + off; off += (bytes + 255) & ~(size_t)255; return p; };
    float*  mem     = (float*)carve((size_t)B * S * H * 4);        // 26.2 MB
    __bf16* encbf   = (__bf16*)carve((size_t)B * S * H * 2);       // 13.1 MB
    __bf16* WencT   = (__bf16*)carve((size_t)H * H * 2);
    __bf16* WoutT   = (__bf16*)carve((size_t)NP_OUT * H * 2);      // 46.1 MB
    __bf16* WihT    = (__bf16*)carve((size_t)4 * H * KP * 2);
    __bf16* WhhT    = (__bf16*)carve((size_t)4 * H * H * 2);
    float*  h       = (float*)carve((size_t)B * H * 4);
    float*  c       = (float*)carve((size_t)B * H * 4);
    float*  ctx     = (float*)carve((size_t)B * H * 4);
    __bf16* hbf     = (__bf16*)carve((size_t)B * H * 2);
    __bf16* xbf     = (__bf16*)carve((size_t)B * KP * 2);
    __bf16* libf    = (__bf16*)carve((size_t)B * H * 2);
    float*  energy  = (float*)carve((size_t)B * S * 4);
    (void)ws_size; (void)n_in; (void)in_sizes; (void)out_size;

    dim3 tb(32, 8);
    // one-time conversions (bf16, transposed weight layouts)
    k_init<<<(B * H + 255) / 256, 256, 0, stream>>>(h0, c0, h, c, ctx);
    k_cvt<<<((B * S * H) + 255) / 256, 256, 0, stream>>>(enc_out, encbf, B * S * H);
    k_cvt_trans<<<dim3(H / 32, H / 32), tb, 0, stream>>>(W_enc, WencT, H, H, H, H);
    k_cvt_trans<<<dim3((4 * H) / 32, KP / 32), tb, 0, stream>>>(W_ih, WihT, E, 4 * H, KP, 4 * H);
    k_cvt_trans<<<dim3((4 * H) / 32, H / 32), tb, 0, stream>>>(W_hh, WhhT, H, 4 * H, H, 4 * H);
    k_cvt_trans<<<dim3((NP_OUT + 31) / 32, H / 32), tb, 0, stream>>>(W_out, WoutT, H, V, H, NP_OUT);
    k_memories<<<(800 * 8 + 7) / 8, 256, 0, stream>>>(encbf, WencT, b_enc, mem);

    const int out_waves = 2 * OUT_GROUPS;                    // 1408 waves
    const int out_blocks = (out_waves + 7) / 8;              // 176 blocks
    for (int t = 0; t < T; ++t) {
        k_reduce<<<(B * KP + 255) / 256, 256, 0, stream>>>(trg, embedding, ctx, W_red, b_red, xbf, t);
        k_h2bf<<<(B * H + 255) / 256, 256, 0, stream>>>(h, hbf);
        k_gates<<<H / 16, 256, 0, stream>>>(xbf, hbf, WihT, WhhT, b_ih, b_hh, c, h);
        k_attn<<<B, 256, 0, stream>>>(h, mem, enc_mask, energy, ctx);
        k_cat<<<(B * H + 255) / 256, 256, 0, stream>>>(h, ctx, W_cat, b_cat, libf);
        k_out<<<out_blocks, 256, 0, stream>>>(libf, WoutT, b_out, out, t);
        k_zoov<<<(B * NOOV + 255) / 256, 256, 0, stream>>>(out, t);
        k_scatter<<<(B * S + 255) / 256, 256, 0, stream>>>(ext_src, energy, out, t);
        k_final<<<(B * VE + 255) / 256, 256, 0, stream>>>(out, t);
    }
}
